// SelfAttention_59227599012166
// MI455X (gfx1250) — matrix operand order repaired and checked by reading, not compiled, not run on hardware
//
#include <hip/hip_runtime.h>

typedef __attribute__((ext_vector_type(16))) _Float16 v16h;
typedef __attribute__((ext_vector_type(8)))  _Float16 v8h;
typedef __attribute__((ext_vector_type(8)))  float    v8f;

#define NB 8
#define NT 2048
#define NH 256
#define NEGV (-1e9f)

static __device__ __forceinline__ v8f wmma16(v16h a, v16h b, v8f c) {
  // D = A(16x32 f16) * B(32x16 f16) + C(16x16 f32)
  return __builtin_amdgcn_wmma_f32_16x16x32_f16(false, a, false, b, (short)0, c,
                                                false, false);
}

// Build a 16-half fragment from two contiguous 8-half (16B) spans.
static __device__ __forceinline__ v16h ld2x8(const _Float16* lo, const _Float16* hi) {
  union { v16h w; v8h h[2]; } u;
  u.h[0] = *(const v8h*)lo;
  u.h[1] = *(const v8h*)hi;
  return u.w;
}

// ---------------- Kernel 1: convert + mask ----------------
__global__ void sa_prep(const float* __restrict__ feats, const float* __restrict__ W,
                        _Float16* __restrict__ feats_h, _Float16* __restrict__ Wh,
                        float* __restrict__ maskNeg) {
  int i = blockIdx.x * blockDim.x + threadIdx.x;
  if (i < NB * NT * NH) feats_h[i] = (_Float16)feats[i];
  if (i < NH * NH)      Wh[i]      = (_Float16)W[i];
  if (i < NB * NT)      maskNeg[i] = (feats[(size_t)i * NH] == 0.0f) ? NEGV : 0.0f;
}

// ---------------- Kernel 2: featsT[b][h][t] (f16), LDS-tiled ----------------
__global__ void sa_transpose(const float* __restrict__ feats, _Float16* __restrict__ featsT) {
  __shared__ float tile[32][33];
  int bx = blockIdx.x;
  int b  = bx >> 9;        // (T/32)*(H/32) = 512 tiles per batch
  int r  = bx & 511;
  int tt = r >> 3;         // 0..63  (t tile)
  int hh = r & 7;          // 0..7   (h tile)
  int tx = threadIdx.x & 31;
  int ty = threadIdx.x >> 5;  // 0..7
#pragma unroll
  for (int i = 0; i < 4; ++i) {
    int row = ty + 8 * i;
    tile[row][tx] = feats[((size_t)(b * NT + tt * 32 + row)) * NH + hh * 32 + tx];
  }
  __syncthreads();
#pragma unroll
  for (int i = 0; i < 4; ++i) {
    int row = ty + 8 * i;
    featsT[((size_t)(b * NH + hh * 32 + row)) * NT + tt * 32 + tx] = (_Float16)tile[tx][row];
  }
}

// ---------------- Kernel 3: proj = feats @ W^T + b  (f16 out) ----------------
__global__ void sa_proj(const _Float16* __restrict__ feats_h, const _Float16* __restrict__ Wh,
                        const float* __restrict__ bias, _Float16* __restrict__ proj_h) {
  int wave = blockIdx.x * (blockDim.x >> 5) + (threadIdx.x >> 5);
  int lane = threadIdx.x & 31;
  int ln = lane & 15, hf = lane >> 4;
  int oo = wave & 15;          // H/16 tiles
  int tt = (wave >> 4) & 127;  // T/16 tiles
  int b  = wave >> 11;

  const _Float16* Arow = feats_h + ((size_t)(b * NT + tt * 16 + ln)) * NH; // A: M=t, K=h
  const _Float16* Brow = Wh + (size_t)(oo * 16 + ln) * NH;                 // B[k=h][n=o]=W[o][h]
  v8f c = {};
#pragma unroll
  for (int k = 0; k < NH; k += 32) {
    v16h a = ld2x8(Arow + k + 8 * hf, Arow + k + 16 + 8 * hf);
    v16h w = ld2x8(Brow + k + 8 * hf, Brow + k + 16 + 8 * hf);
    c = wmma16(a, w, c);
  }
  float bb = bias[oo * 16 + ln];
#pragma unroll
  for (int r2 = 0; r2 < 8; ++r2) {
    proj_h[((size_t)(b * NT + tt * 16 + r2 + 8 * hf)) * NH + oo * 16 + ln] =
        (_Float16)(c[r2] + bb);
  }
}

// ---------------- Kernel 4: flash attention, one wave per 16-query tile ----------------
__global__ void sa_attn(const _Float16* __restrict__ feats_h, const _Float16* __restrict__ featsT,
                        const _Float16* __restrict__ proj_h, const float* __restrict__ maskNeg,
                        float* __restrict__ partials) {
  int wave = blockIdx.x * (blockDim.x >> 5) + (threadIdx.x >> 5);
  int lane = threadIdx.x & 31;
  int ln = lane & 15, hf = lane >> 4;
  int qt = wave & 127;   // T/16 query tiles
  int b  = wave >> 7;

  // Preload Q as B-operand of S^T = K * Q^T :  B[k=h][n=q] = proj[q][h]
  const _Float16* Qrow = proj_h + ((size_t)(b * NT + qt * 16 + ln)) * NH;
  v16h qf[8];
#pragma unroll
  for (int j = 0; j < 8; ++j)
    qf[j] = ld2x8(Qrow + 32 * j + 8 * hf, Qrow + 32 * j + 16 + 8 * hf);

  v8f oacc[16];
#pragma unroll
  for (int j = 0; j < 16; ++j) oacc[j] = (v8f){};
  float m = -3.0e38f, l = 0.0f;

  const _Float16* Kb   = feats_h + (size_t)b * NT * NH;
  const _Float16* Vt   = featsT + (size_t)b * NH * NT;
  const float*    mrow = maskNeg + (size_t)b * NT;

  for (int s0 = 0; s0 < NT; s0 += 32) {
    // S^T for 32 keys: c0 = keys s0..s0+15, c1 = keys s0+16..s0+31
    const _Float16* K0 = Kb + (size_t)(s0 + ln) * NH;   // A: M=s (lane), K=h
    const _Float16* K1 = K0 + (size_t)16 * NH;
    v8f c0 = {}, c1 = {};
#pragma unroll
    for (int j = 0; j < 8; ++j) {
      v16h a0 = ld2x8(K0 + 32 * j + 8 * hf, K0 + 32 * j + 16 + 8 * hf);
      c0 = wmma16(a0, qf[j], c0);
    }
#pragma unroll
    for (int j = 0; j < 8; ++j) {
      v16h a1 = ld2x8(K1 + 32 * j + 8 * hf, K1 + 32 * j + 16 + 8 * hf);
      c1 = wmma16(a1, qf[j], c1);
    }
    // lane holds S[q=ln][s = s0 + r + 8*hf] in c0[r], +16 in c1[r]
    float p0[8], p1[8];
    float mx = -3.0e38f;
#pragma unroll
    for (int r2 = 0; r2 < 8; ++r2) {
      float v0 = c0[r2] + mrow[s0 + r2 + 8 * hf];
      float v1 = c1[r2] + mrow[s0 + 16 + r2 + 8 * hf];
      p0[r2] = v0; p1[r2] = v1;
      mx = fmaxf(mx, fmaxf(v0, v1));
    }
    mx = fmaxf(mx, __shfl_xor(mx, 16, 32));   // combine lane pair (q, q+16)
    float mnew  = fmaxf(m, mx);
    float scale = __expf(m - mnew);
    float rs = 0.0f;
#pragma unroll
    for (int r2 = 0; r2 < 8; ++r2) {
      p0[r2] = __expf(p0[r2] - mnew);
      p1[r2] = __expf(p1[r2] - mnew);
      rs += p0[r2] + p1[r2];
    }
    rs += __shfl_xor(rs, 16, 32);
    l = l * scale + rs;
    m = mnew;
    // Rescale O accumulators: row q = r + 8*hf needs scale of query (r+8*hf)
    float sc[8];
#pragma unroll
    for (int r2 = 0; r2 < 8; ++r2) sc[r2] = __shfl(scale, r2 + 8 * hf, 32);
#pragma unroll
    for (int j = 0; j < 16; ++j)
#pragma unroll
      for (int r2 = 0; r2 < 8; ++r2) oacc[j][r2] *= sc[r2];
    // P as A-operand (M=q, K=32 keys): S^T C-layout == A-layout, register-local.
    v16h pa;
#pragma unroll
    for (int r2 = 0; r2 < 8; ++r2) {
      pa[r2]     = (_Float16)p0[r2];
      pa[8 + r2] = (_Float16)p1[r2];
    }
    // O += P * V : B[k=s][n=h] = featsT[h][s] (row-major friendly)
#pragma unroll
    for (int j = 0; j < 16; ++j) {
      const _Float16* Vr = Vt + (size_t)(16 * j + ln) * NT + s0 + 8 * hf;
      v16h vb = ld2x8(Vr, Vr + 16);
      oacc[j] = wmma16(pa, vb, oacc[j]);
    }
  }

  // Divide by l per O-row and sum over the 16 queries of this tile.
  float li = 1.0f / l;
  float lr[8];
#pragma unroll
  for (int r2 = 0; r2 < 8; ++r2) lr[r2] = __shfl(li, r2 + 8 * hf, 32);
#pragma unroll
  for (int j = 0; j < 16; ++j) {
    float s = 0.0f;
#pragma unroll
    for (int r2 = 0; r2 < 8; ++r2) s += oacc[j][r2] * lr[r2];
    s += __shfl_xor(s, 16, 32);               // rows 0..7 + rows 8..15
    if (lane < 16)
      partials[(size_t)wave * NH + 16 * j + ln] = s;
  }
}

// ---------------- Kernel 5: deterministic reduction over query tiles ----------------
__global__ void sa_reduce(const float* __restrict__ partials, float* __restrict__ out) {
  int i = blockIdx.x * blockDim.x + threadIdx.x;
  if (i >= NB * NH) return;
  int b = i >> 8, h = i & 255;
  float s = 0.0f;
  for (int q = 0; q < NT / 16; ++q)
    s += partials[((size_t)(b * (NT / 16) + q)) * NH + h];
  out[i] = s;
}

extern "C" void kernel_launch(void* const* d_in, const int* in_sizes, int n_in,
                              void* d_out, int out_size, void* d_ws, size_t ws_size,
                              hipStream_t stream) {
  (void)in_sizes; (void)n_in; (void)out_size; (void)ws_size;
  const float* feats = (const float*)d_in[0];
  const float* W     = (const float*)d_in[1];
  const float* bias  = (const float*)d_in[2];
  float* out = (float*)d_out;

  char* ws = (char*)d_ws;
  const size_t SZH = (size_t)NB * NT * NH * sizeof(_Float16);  // 8 MiB
  _Float16* feats_h = (_Float16*)(ws);
  _Float16* featsT  = (_Float16*)(ws + SZH);
  _Float16* proj_h  = (_Float16*)(ws + 2 * SZH);
  float*    maskNeg = (float*)(ws + 3 * SZH);
  _Float16* Wh      = (_Float16*)(ws + 3 * SZH + 65536);
  float*    partials= (float*)(ws + 3 * SZH + 65536 + 131072);

  sa_prep<<<(NB * NT * NH + 255) / 256, 256, 0, stream>>>(feats, W, feats_h, Wh, maskNeg);
  sa_transpose<<<NB * (NT / 32) * (NH / 32), 256, 0, stream>>>(feats, featsT);
  sa_proj<<<NB * (NT / 16) * (NH / 16) / 8, 256, 0, stream>>>(feats_h, Wh, bias, proj_h);
  sa_attn<<<NB * (NT / 16) / 8, 256, 0, stream>>>(feats_h, featsT, proj_h, maskNeg, partials);
  sa_reduce<<<(NB * NH + 255) / 256, 256, 0, stream>>>(partials, out);
}
